// AttentiveQuantizer_31044023615833
// MI455X (gfx1250) — compile-verified, matrix operand order checked by reading, not run on hardware
//
#include <hip/hip_runtime.h>
#include <hip/hip_bf16.h>

typedef __attribute__((ext_vector_type(16))) __bf16 v16bf;
typedef __attribute__((ext_vector_type(8)))  float  v8f;
typedef __attribute__((ext_vector_type(4)))  unsigned su4;
typedef __attribute__((ext_vector_type(8)))  unsigned su8;

#define CIN   512
#define COUT  512
#define KCB   1024
#define HWSZ  4096
#define NHW   65536

// ---------------------------------------------------------------- WMMA core
__device__ __forceinline__ v8f wmma_bf16(v16bf a, v16bf b, v8f c) {
  // D = A(16x32) * B(32x16) + C, emits v_wmma_f32_16x16x32_bf16
  return __builtin_amdgcn_wmma_f32_16x16x32_bf16(false, a, false, b, (short)0, c,
                                                 false, false);
}

// ---------------------------------------------------------------- hash RNG
__device__ __forceinline__ unsigned pcg(unsigned x) {
  x = x * 747796405u + 2891336453u;
  unsigned w = ((x >> ((x >> 28u) + 4u)) ^ x) * 277803737u;
  return (w >> 22u) ^ w;
}
__device__ __forceinline__ float u01(unsigned a, unsigned b, unsigned salt) {
  unsigned h = pcg(a ^ pcg(b ^ pcg(salt)));
  return (float)(h >> 8) * (1.0f / 16777216.0f);
}

// ---------------------------------------------------------------- f32 -> bf16 pack
__global__ void pack_bf16_kernel(const float* __restrict__ src,
                                 __bf16* __restrict__ dst, int n) {
  int i = blockIdx.x * blockDim.x + threadIdx.x;
  if (i < n) dst[i] = (__bf16)src[i];
}

// ---------------------------------------------------------------- generic GEMM
// C(MxN) = A(MxK) * B(NxK)^T ; A,B bf16 row-major; out bf16 or f32.
// block = 256 threads (8 waves), tile = 128 x 64, K step 32.
__global__ void gemm_bf16_nt(const __bf16* __restrict__ A,
                             const __bf16* __restrict__ B,
                             void* __restrict__ Cout,
                             int Kdim, int Ndim, int out_f32, float scale) {
  __shared__ __bf16 As[128][32];
  const int t = threadIdx.x;
  const int w = t >> 5, lane = t & 31;
  const int hi = lane >> 4, nlo = lane & 15;
  const int m0 = blockIdx.x * 128;
  const int n0 = blockIdx.y * 64;
  v8f acc[4] = {};
  const int srow = t >> 1, sseg = t & 1;
  for (int c0 = 0; c0 < Kdim; c0 += 32) {
    // cooperative stage of 128x32 bf16 A tile
    v16bf av = *(const v16bf*)(A + (size_t)(m0 + srow) * Kdim + c0 + sseg * 16);
    *(v16bf*)&As[srow][sseg * 16] = av;
    __builtin_prefetch(A + (size_t)(m0 + srow) * Kdim + c0 + 32, 0, 3);
    __syncthreads();
    v16bf af = *(const v16bf*)&As[w * 16 + nlo][hi * 16];
#pragma unroll
    for (int s = 0; s < 4; ++s) {
      int col = n0 + s * 16 + nlo;
      v16bf bf = *(const v16bf*)(B + (size_t)col * Kdim + c0 + hi * 16);
      acc[s] = wmma_bf16(af, bf, acc[s]);
    }
    __syncthreads();
  }
#pragma unroll
  for (int s = 0; s < 4; ++s) {
#pragma unroll
    for (int j = 0; j < 8; ++j) {
      int rg = m0 + w * 16 + j + 8 * hi;   // C layout: VGPR j -> M=j (+8 for hi lanes)
      int cg = n0 + s * 16 + nlo;
      float val = acc[s][j] * scale;
      if (out_f32) ((float*)Cout)[(size_t)rg * Ndim + cg] = val;
      else ((__bf16*)Cout)[(size_t)rg * Ndim + cg] = (__bf16)val;
    }
  }
}

// ---------------------------------------------------------------- q = latent . wq^T
// A staged from NCHW f32 latent with LDS transpose + bf16 convert.
__global__ void gemm_latent_q(const float* __restrict__ latent,
                              const __bf16* __restrict__ Wq,
                              __bf16* __restrict__ Q) {
  __shared__ __bf16 As[128][32];
  const int t = threadIdx.x;
  const int w = t >> 5, lane = t & 31;
  const int hi = lane >> 4, nlo = lane & 15;
  const int m0 = blockIdx.x * 128;          // 128 contiguous nhw rows (same n)
  const int n  = m0 >> 12;
  const int hw0 = m0 & (HWSZ - 1);
  const int n0 = blockIdx.y * 64;
  v8f acc[4] = {};
  const int cr = t >> 3, seg = t & 7;       // 32 channels x 8 segments of 16 hw
  for (int c0 = 0; c0 < CIN; c0 += 32) {
    const float4* s4 =
        (const float4*)(latent + ((size_t)n * CIN + c0 + cr) * HWSZ + hw0 + seg * 16);
    __builtin_prefetch((const float*)s4 + HWSZ * 32, 0, 3);
#pragma unroll
    for (int q4 = 0; q4 < 4; ++q4) {
      float4 vv = s4[q4];
      int rb = seg * 16 + q4 * 4;
      As[rb + 0][cr] = (__bf16)vv.x;        // transposed store: A[row][c]
      As[rb + 1][cr] = (__bf16)vv.y;
      As[rb + 2][cr] = (__bf16)vv.z;
      As[rb + 3][cr] = (__bf16)vv.w;
    }
    __syncthreads();
    v16bf af = *(const v16bf*)&As[w * 16 + nlo][hi * 16];
#pragma unroll
    for (int s = 0; s < 4; ++s) {
      int col = n0 + s * 16 + nlo;
      v16bf bf = *(const v16bf*)(Wq + (size_t)col * CIN + c0 + hi * 16);
      acc[s] = wmma_bf16(af, bf, acc[s]);
    }
    __syncthreads();
  }
#pragma unroll
  for (int s = 0; s < 4; ++s) {
#pragma unroll
    for (int j = 0; j < 8; ++j) {
      int rg = m0 + w * 16 + j + 8 * hi;
      int cg = n0 + s * 16 + nlo;
      Q[(size_t)rg * COUT + cg] = (__bf16)acc[s][j];
    }
  }
}

// ---------------------------------------------------------------- logit GEMM + stats
// block = 256 threads, 16 rows x full K=1024 cols.
// A tile (16x32 bf16) staged into LDS by the Tensor Data Mover (wave 0 issues a
// real D#: count=1, data_size=2B, tensor 65536x512, tile 16x32, stride 512).
// Fused argmax + binCount f32 atomics in the epilogue.
__global__ void gemm_logit_stats(const __bf16* __restrict__ Q,
                                 const __bf16* __restrict__ Km,
                                 float* __restrict__ logit,
                                 float* __restrict__ trueCode,
                                 float* __restrict__ binCount) {
  __shared__ __bf16 As[16][32];
  __shared__ float wvred[8][16];
  __shared__ int   wcred[8][16];
  const int t = threadIdx.x;
  const int w = t >> 5, lane = t & 31;
  const int hi = lane >> 4, nlo = lane & 15;
  const int r0 = blockIdx.x * 16;
  const int n = r0 >> 12;
  v8f acc[8] = {};
  const unsigned lds_off = (unsigned)(size_t)(&As[0][0]);
  for (int c0 = 0; c0 < COUT; c0 += 32) {
    if (w == 0) {
      // Tensor DMA: load 16x32 bf16 tile of Q starting at (r0, c0) into As.
      unsigned long long ga =
          (unsigned long long)(size_t)Q + ((size_t)r0 * COUT + c0) * 2ull;
      su4 g0;
      g0[0] = 1u;                                   // count=1, user descriptor
      g0[1] = lds_off;                              // lds_addr (bytes)
      g0[2] = (unsigned)ga;                         // global_addr[31:0]
      g0[3] = (unsigned)(ga >> 32) | 0x80000000u;   // global_addr[56:32] | type=2
      su8 g1;
      g1[0] = 0x10000u;                             // data_size=1 (2 bytes)
      g1[1] = ((unsigned)COUT & 0xFFFFu) << 16;     // tensor_dim0[15:0]=512
      g1[2] = ((unsigned)COUT >> 16) |
              (((unsigned)NHW & 0xFFFFu) << 16);    // tensor_dim0 hi | tensor_dim1 lo
      g1[3] = ((unsigned)NHW >> 16) | (32u << 16);  // tensor_dim1 hi | tile_dim0=32
      g1[4] = 16u;                                  // tile_dim1=16, tile_dim2=0
      g1[5] = (unsigned)COUT;                       // tensor_dim0_stride=512
      g1[6] = 0u;
      g1[7] = 0u;                                   // tensor_dim1_stride unused
      asm volatile("tensor_load_to_lds %0, %1" :: "s"(g0), "s"(g1) : "memory");
      __builtin_amdgcn_s_wait_tensorcnt(0);
    }
    __builtin_prefetch(Q + (size_t)r0 * COUT + c0 + 32, 0, 3);
    __syncthreads();
    v16bf af = *(const v16bf*)&As[nlo][hi * 16];
#pragma unroll
    for (int s = 0; s < 8; ++s) {
      int col = w * 128 + s * 16 + nlo;     // wave owns a 128-col strip
      v16bf bf = *(const v16bf*)(Km + (size_t)col * COUT + c0 + hi * 16);
      acc[s] = wmma_bf16(af, bf, acc[s]);
    }
    __syncthreads();
  }
  const float inv_scale = 0.044194173824159216f;  // 1/sqrt(512)
#pragma unroll
  for (int j = 0; j < 8; ++j) {
    const int rl = j + 8 * hi;
    const int row = r0 + rl;
    float bv = -3.4e38f; int bc = 0;
#pragma unroll
    for (int s = 0; s < 8; ++s) {
      float val = acc[s][j] * inv_scale;
      int cg = w * 128 + s * 16 + nlo;
      logit[(size_t)row * KCB + cg] = val;
      if (val > bv) { bv = val; bc = cg; }
    }
    // reduce over the 16 lanes of this half-wave (same M row, different N)
    for (int mlane = 1; mlane < 16; mlane <<= 1) {
      float ov = __shfl_xor(bv, mlane, 32);
      int   oc = __shfl_xor(bc, mlane, 32);
      if (ov > bv || (ov == bv && oc < bc)) { bv = ov; bc = oc; }
    }
    if (nlo == 0) { wvred[w][rl] = bv; wcred[w][rl] = bc; }
  }
  __syncthreads();
  if (t < 16) {
    float bv = wvred[0][t]; int bc = wcred[0][t];
#pragma unroll
    for (int ww = 1; ww < 8; ++ww) {
      float ov = wvred[ww][t]; int oc = wcred[ww][t];
      if (ov > bv || (ov == bv && oc < bc)) { bv = ov; bc = oc; }
    }
    trueCode[r0 + t] = (float)bc;
    atomicAdd(&binCount[n * KCB + bc], 1.0f);
  }
}

// ---------------------------------------------------------------- finalize
// block = 256 threads handles 64 consecutive hw rows.  Each wave: 8 rows with
// coalesced logit reads + wave32 shfl reductions.  Then a cooperative,
// fully-coalesced NCHW scatter of v[code] (128B contiguous per wave-store).
__global__ void finalize_kernel(const float* __restrict__ logit,
                                const float* __restrict__ trueCode,
                                const float* __restrict__ binCount,
                                const float* __restrict__ Vmat,
                                const int* __restrict__ tptr,
                                float* __restrict__ quant,
                                float* __restrict__ code_out,
                                float* __restrict__ freq_out) {
  __shared__ int scode[64];
  const int t = threadIdx.x;
  const int w = t >> 5, lane = t & 31;
  const int rbase = blockIdx.x * 64;     // 64 rows, all in the same image n
  const int n = rbase >> 12;
  const int hw0 = rbase & (HWSZ - 1);
  const float temp = (float)tptr[0];
  for (int rr = 0; rr < 8; ++rr) {
    const int rl = w * 8 + rr;
    const int r = rbase + rl;
    float L[32];
#pragma unroll
    for (int i = 0; i < 32; ++i) L[i] = logit[(size_t)r * KCB + i * 32 + lane];
    // row max
    float m = -3.4e38f;
#pragma unroll
    for (int i = 0; i < 32; ++i) m = fmaxf(m, L[i]);
    for (int mk = 16; mk > 0; mk >>= 1) m = fmaxf(m, __shfl_xor(m, mk, 32));
    // row sum of exp
    float s = 0.f;
#pragma unroll
    for (int i = 0; i < 32; ++i) s += __expf(L[i] - m);
    for (int mk = 16; mk > 0; mk >>= 1) s += __shfl_xor(s, mk, 32);
    const float inv_s = 1.0f / s;
    const int tc = (int)trueCode[r];
    const float freq = binCount[n * KCB + tc];
    const bool drop =
        u01((unsigned)r, 0x9e3779b9u, 0xD00DFEEDu) < freq * (1.0f / 4096.0f);
    // masked gumbel argmax
    float bv = -3.4e38f; int bc = 0;
#pragma unroll
    for (int i = 0; i < 32; ++i) {
      int c = i * 32 + lane;
      float prob = __expf(L[i] - m) * inv_s;
      if (prob < (1.0f / 1024.0f)) prob = 0.0f;
      bool m1 = u01((unsigned)r, (unsigned)c, 0xA5A5A5A5u) < prob;  // bernoulli
      bool sc2 = (c == tc) && drop;                                  // scatter mask
      float ml = (m1 | sc2) ? -1.0e9f : L[i];
      float u = u01((unsigned)r, (unsigned)c, 0xB16B00B5u);
      u = u * 0.99999988f + 5.0e-8f;                                 // keep in (0,1)
      float g = -__logf(-__logf(u));                                 // gumbel
      float y = (ml + g) / temp;
      if (y > bv) { bv = y; bc = c; }
    }
    for (int mk = 1; mk < 32; mk <<= 1) {
      float ov = __shfl_xor(bv, mk, 32);
      int   oc = __shfl_xor(bc, mk, 32);
      if (ov > bv || (ov == bv && oc < bc)) { bv = ov; bc = oc; }
    }
    if (lane == 0) {
      scode[rl] = bc;
      code_out[r] = (float)bc;
      freq_out[r] = freq;
    }
  }
  __syncthreads();
  // quantized[n, c, hw0 + hwl] = v[code[hwl], c]; per-wave stores are 32
  // consecutive floats (128B) in hw -> fully coalesced.
  const int hwl = t & 63, csub = t >> 6;
  const int code = scode[hwl];
  for (int cc = 0; cc < COUT; cc += 4) {
    int c = cc + csub;
    quant[((size_t)n * COUT + c) * HWSZ + hw0 + hwl] =
        Vmat[(size_t)code * COUT + c];
  }
}

// ---------------------------------------------------------------- launch
extern "C" void kernel_launch(void* const* d_in, const int* in_sizes, int n_in,
                              void* d_out, int out_size, void* d_ws, size_t ws_size,
                              hipStream_t stream) {
  (void)in_sizes; (void)n_in; (void)out_size; (void)ws_size;
  const float* latent   = (const float*)d_in[0];
  const float* codebook = (const float*)d_in[1];
  const float* wq       = (const float*)d_in[2];
  const float* wk       = (const float*)d_in[3];
  const float* wv       = (const float*)d_in[4];
  const int*   tptr     = (const int*)d_in[5];

  char* ws = (char*)d_ws;
  __bf16* q_bf    = (__bf16*)(ws);                    // 65536*512*2 = 64 MB
  __bf16* kmat_bf = (__bf16*)(ws + 67108864ull);      // 1 MB
  __bf16* cb_bf   = (__bf16*)(ws + 68157440ull);      // 1 MB
  __bf16* wq_bf   = (__bf16*)(ws + 69206016ull);      // 0.5 MB
  __bf16* wk_bf   = (__bf16*)(ws + 69730304ull);      // 0.5 MB
  __bf16* wv_bf   = (__bf16*)(ws + 70254592ull);      // 0.5 MB
  float*  v_f32   = (float*) (ws + 70778880ull);      // 2 MB

  float* out       = (float*)d_out;                   // outputs concatenated flat:
  float* quant     = out;                             // (16,512,64,64)
  float* code_out  = out + 33554432ull;               // (16,64,64)
  float* logit     = out + 33619968ull;               // (16,64,64,1024)
  float* trueCode  = out + 100728832ull;              // (16,64,64)
  float* freq_out  = out + 100794368ull;              // (16,64,64)
  float* binCount  = out + 100859904ull;              // (16,1024)

  // 1) pack weights/codebook to bf16
  pack_bf16_kernel<<<1024, 256, 0, stream>>>(wq, wq_bf, COUT * CIN);
  pack_bf16_kernel<<<1024, 256, 0, stream>>>(wk, wk_bf, COUT * COUT);
  pack_bf16_kernel<<<1024, 256, 0, stream>>>(wv, wv_bf, COUT * COUT);
  pack_bf16_kernel<<<2048, 256, 0, stream>>>(codebook, cb_bf, KCB * CIN);

  // 2) kmat = codebook . wk^T (bf16 out), v = codebook . wv^T (f32 out)
  dim3 gkv(KCB / 128, COUT / 64);
  gemm_bf16_nt<<<gkv, 256, 0, stream>>>(cb_bf, wk_bf, (void*)kmat_bf, CIN, COUT, 0, 1.0f);
  gemm_bf16_nt<<<gkv, 256, 0, stream>>>(cb_bf, wv_bf, (void*)v_f32,   CIN, COUT, 1, 1.0f);

  // 3) q = latent . wq^T (staged from NCHW, bf16 out)
  dim3 gq(NHW / 128, COUT / 64);
  gemm_latent_q<<<gq, 256, 0, stream>>>(latent, wq_bf, q_bf);

  // 4) logit GEMM (TDM-staged A) + argmax/binCount (binCount zeroed first)
  hipMemsetAsync(binCount, 0, 16 * KCB * sizeof(float), stream);
  gemm_logit_stats<<<NHW / 16, 256, 0, stream>>>(q_bf, kmat_bf, logit, trueCode, binCount);

  // 5) per-row softmax / masks / gumbel argmax / coalesced gather of v[code]
  finalize_kernel<<<NHW / 64, 256, 0, stream>>>(logit, trueCode, binCount, v_f32, tptr,
                                                quant, code_out, freq_out);
}
